// Decoder_84464826843223
// MI455X (gfx1250) — matrix operand order repaired and checked by reading, not compiled, not run on hardware
//
#include <hip/hip_runtime.h>
#include <hip/hip_bf16.h>
#include <cstdint>

// ---------------------------------------------------------------------------
// Problem constants (from reference)
// ---------------------------------------------------------------------------
#define BB    32          // batch
#define TT    800         // time steps
#define DIN   576         // layer-1 input dim
#define UU    512         // LSTM units
#define GG    2048        // 4*UU gate width
#define H2    1024        // bilstm output width
#define NMEL  80
#define MROWS (BB*TT)     // 25600 GEMM rows

#define NWG       4       // workgroups per direction in the scan
#define SCAN_THREADS 256  // 8 waves of 32
#define HSTATE    (BB*UU) // 16384 elems per h buffer

typedef __bf16 bf16;
typedef __attribute__((ext_vector_type(16))) __bf16 v16bf;
typedef __attribute__((ext_vector_type(8)))  float  v8f;
typedef __attribute__((ext_vector_type(4)))  unsigned int u32x4;

union Frag { v16bf v; u32x4 u[2]; };

static __device__ inline v8f zf8() {
  v8f z;
#pragma unroll
  for (int i = 0; i < 8; ++i) z[i] = 0.f;
  return z;
}

static __device__ inline v8f wmma_bf16(v16bf a, v16bf b, v8f c) {
  // D = A(16x32 bf16) * B(32x16 bf16) + C(16x16 f32)
  return __builtin_amdgcn_wmma_f32_16x16x32_bf16(
      /*neg_a=*/false, a, /*neg_b=*/false, b,
      /*c_mod=*/(short)0, c, /*reuse_a=*/false, /*reuse_b=*/false);
}

// A fragment: row-major bf16 [*, lda], tile at (m0, k0).
// ISA 16-bit A 16x32 layout: lanes 0-15 row M=lane, K={0..7,16..23};
// lanes 16-31 row M=lane-16, K={8..15,24..31}.  Two b128 loads per lane.
static __device__ inline v16bf load_a_frag(const bf16* A, int lda, int m0,
                                           int k0, int lane) {
  int mloc = lane & 15;
  int kb = (lane >> 4) << 3;              // 0 or 8
  const bf16* p = A + (size_t)(m0 + mloc) * lda + k0 + kb;
  Frag f;
  f.u[0] = *(const u32x4*)(p);            // K = k0+kb   .. +7
  f.u[1] = *(const u32x4*)(p + 16);       // K = k0+kb+16.. +23
  return f.v;
}

// B fragment from pre-packed weights: 16 contiguous bf16 per lane per
// (n-tile, k-tile).  Layout produced by pack_b_kernel below.
static __device__ inline v16bf load_b_frag(const bf16* Bp, int nt, int kt,
                                           int ktiles, int lane) {
  const bf16* p = Bp + (((size_t)nt * ktiles + kt) * 32 + lane) * 16;
  Frag f;
  f.u[0] = *(const u32x4*)(p);
  f.u[1] = *(const u32x4*)(p + 8);
  return f.v;
}

// ---------------------------------------------------------------------------
// Elementwise / packing utility kernels
// ---------------------------------------------------------------------------
__global__ void f32_to_bf16_kernel(const float* __restrict__ src,
                                   bf16* __restrict__ dst, long n) {
  long i = (long)blockIdx.x * blockDim.x + threadIdx.x;
  long stride = (long)gridDim.x * blockDim.x;
  for (; i < n; i += stride) dst[i] = (bf16)src[i];
}

// Pack row-major fp32 W[K,N] into WMMA B-fragment order (bf16):
// ISA 16-bit B 32x16 layout: lane = ((kin>>3)&1)*16 + n%16 holds K values
// (kin&7)|((kin>>4)<<3) -> element.  out[((nt*ktiles+kt)*32+lane)*16 + e]
__global__ void pack_b_kernel(const float* __restrict__ W,
                              bf16* __restrict__ out, int K, int N) {
  long idx = (long)blockIdx.x * blockDim.x + threadIdx.x;
  if (idx >= (long)K * N) return;
  int k = (int)(idx / N), n = (int)(idx % N);
  int kt = k >> 5, kin = k & 31;
  int nt = n >> 4, nl = n & 15;
  int lane = (((kin >> 3) & 1) << 4) | nl;
  int e = (kin & 7) | ((kin >> 4) << 3);
  int ktiles = K >> 5;
  out[(((size_t)nt * ktiles + kt) * 32 + lane) * 16 + e] = (bf16)W[idx];
}

__global__ void zero_u32_kernel(unsigned int* __restrict__ p, long nwords) {
  long i = (long)blockIdx.x * blockDim.x + threadIdx.x;
  long stride = (long)gridDim.x * blockDim.x;
  for (; i < nwords; i += stride) p[i] = 0u;
}

// ---------------------------------------------------------------------------
// Register-blocked WMMA GEMM:
//   C[M,N](fp32) = A[M,K](bf16 row-major) * Bpacked + bias[N]
// Each wave computes a 32 x (16*NT) macro-tile (2 M-tiles x NT N-tiles):
// A fragments reused NT times, B fragments reused 2x ->
// (2+NT)/(2*NT) fragment loads per WMMA (0.75 for NT=4) instead of 2.0.
// 8 waves per block.  M must be a multiple of 32, N of 16*NT.
// remapT != 0: store row r=b*T+t remapped to t*BB+b (time-major xg layout).
// ---------------------------------------------------------------------------
template <int NT>
__global__ void __launch_bounds__(256)
wmma_gemm_blocked_kernel(const bf16* __restrict__ A,
                         const bf16* __restrict__ Bp,
                         const float* __restrict__ bias,
                         float* __restrict__ C,
                         int M, int K, int N, int remapT) {
  int lane = threadIdx.x & 31;
  int wave = threadIdx.x >> 5;
  int nMacroN = N / (16 * NT);
  long macro = (long)blockIdx.x * 8 + wave;
  long total = (long)(M >> 5) * nMacroN;
  if (macro >= total) return;
  int mBase = (int)(macro / nMacroN) << 5;
  int nBase = (int)(macro % nMacroN) * (16 * NT);
  int ktiles = K >> 5;

  v8f acc[2][NT];
#pragma unroll
  for (int mt = 0; mt < 2; ++mt)
#pragma unroll
    for (int j = 0; j < NT; ++j) acc[mt][j] = zf8();

  for (int kt = 0; kt < ktiles; ++kt) {
    v16bf a0 = load_a_frag(A, K, mBase,      kt << 5, lane);
    v16bf a1 = load_a_frag(A, K, mBase + 16, kt << 5, lane);
#pragma unroll
    for (int j = 0; j < NT; ++j) {
      v16bf b = load_b_frag(Bp, (nBase >> 4) + j, kt, ktiles, lane);
      acc[0][j] = wmma_bf16(a0, b, acc[0][j]);
      acc[1][j] = wmma_bf16(a1, b, acc[1][j]);
    }
  }

  int hi8 = (lane >> 4) << 3;
#pragma unroll
  for (int j = 0; j < NT; ++j) {
    int n = nBase + (j << 4) + (lane & 15);
    float bv = bias ? bias[n] : 0.f;
#pragma unroll
    for (int mt = 0; mt < 2; ++mt) {
      int mb = mBase + (mt << 4) + hi8;     // C/D layout: M = e + hi8
#pragma unroll
      for (int e = 0; e < 8; ++e) {
        int m = mb + e;
        long row = m;
        if (remapT) {                       // r = b*T + t  ->  t*BB + b
          int t = m % remapT;
          int b_ = m / remapT;
          row = (long)t * BB + b_;
        }
        C[row * (long)N + n] = acc[mt][j][e] + bv;
      }
    }
  }
}

// ---------------------------------------------------------------------------
// Persistent LSTM scan. grid = 2*NWG blocks of 256 threads.
//   dir 0: forward, dir 1: backward (reversed time, Keras mask semantics).
// Each wave owns 16 units x all 32 batch rows; c/h state stays in registers.
// h (bf16) is double-buffered in global per direction; one atomic generation
// barrier per step synchronizes the NWG workgroups of a direction.
// __launch_bounds__(..., 1): this kernel is persistent (one WG per WGP) --
// give it the full VGPR budget so Wr fragments are never spilled to scratch.
// ---------------------------------------------------------------------------
static __device__ inline float sigf(float x) {
  return 1.f / (1.f + __expf(-x));
}

__global__ void __launch_bounds__(SCAN_THREADS, 1)
lstm_scan_kernel(const float* __restrict__ xg0, const float* __restrict__ xg1,
                 const bf16* __restrict__ wr0, const bf16* __restrict__ wr1,
                 bf16* __restrict__ hbuf,      // [2 dirs][2 bufs][BB*UU]
                 bf16* __restrict__ out,       // [BB*TT, H2]
                 const int* __restrict__ mel_len,
                 int* __restrict__ bars) {     // [2 dirs][2] = cnt, gen
  int dir  = blockIdx.x / NWG;
  int wg   = blockIdx.x % NWG;
  int lane = threadIdx.x & 31;
  int wave = threadIdx.x >> 5;

  const float* xg  = dir ? xg1 : xg0;
  const bf16*  wrp = dir ? wr1 : wr0;
  bf16* hb0 = hbuf + (size_t)dir * 2 * HSTATE;
  bf16* hb1 = hb0 + HSTATE;
  int* cnt = bars + dir * 2;
  int* gen = cnt + 1;
  const bool rev = (dir == 1);
  const int u0 = wg * 128 + wave * 16;     // unit base for this wave
  const int coloff = dir * UU;

  const int nloc  = lane & 15;
  const int nunit = u0 + nloc;
  const int hi8   = (lane >> 4) << 3;

  float cst[2][8], hst[2][8];
  int ml[2][8];
#pragma unroll
  for (int mt = 0; mt < 2; ++mt) {
#pragma unroll
    for (int e = 0; e < 8; ++e) {
      cst[mt][e] = 0.f;
      hst[mt][e] = 0.f;
      ml[mt][e] = mel_len[mt * 16 + hi8 + e];
    }
  }

  for (int s = 0; s < TT; ++s) {
    const int t = rev ? (TT - 1 - s) : s;
    const bf16* hread  = (s & 1) ? hb1 : hb0;
    bf16*       hwrite = (s & 1) ? hb0 : hb1;

    // Opaque zero offset, re-laundered every step: the Wr fragment loads
    // cannot be proven loop-invariant over the time loop (so they are not
    // hoisted into scratch spills), while the base pointer keeps its global
    // address space (loads stay global_load_b128, LOADcnt only -- no flat
    // aperture checks / DScnt coupling).  Weights stream from L2 (8 MB bf16
    // for all 4 directions -- resident in the 192 MB L2).
    size_t woff = 0;
    asm volatile("" : "+s"(woff));
    const bf16* wrp_s = wrp + woff;

    // prefetch this step's xg slab (contiguous 256KB in time-major layout)
    __builtin_prefetch(xg + (size_t)t * BB * GG, 0, 1);

    v8f acc[4][2];
#pragma unroll
    for (int g = 0; g < 4; ++g) { acc[g][0] = zf8(); acc[g][1] = zf8(); }

    for (int kt = 0; kt < (UU >> 5); ++kt) {
      v16bf a0 = load_a_frag(hread, UU, 0,  kt << 5, lane);
      v16bf a1 = load_a_frag(hread, UU, 16, kt << 5, lane);
#pragma unroll
      for (int g = 0; g < 4; ++g) {
        int nt = g * (UU >> 4) + (u0 >> 4);       // gate col tile in [0,128)
        v16bf b = load_b_frag(wrp_s, nt, kt, UU >> 5, lane);
        acc[g][0] = wmma_bf16(a0, b, acc[g][0]);
        acc[g][1] = wmma_bf16(a1, b, acc[g][1]);
      }
    }

    // gates + state update (state in registers; Keras masked-carry semantics)
#pragma unroll
    for (int mt = 0; mt < 2; ++mt) {
#pragma unroll
      for (int e = 0; e < 8; ++e) {
        int m = mt * 16 + hi8 + e;
        const float* xr = xg + ((size_t)t * BB + m) * GG;
        float zi = acc[0][mt][e] + xr[0 * UU + nunit];
        float zf = acc[1][mt][e] + xr[1 * UU + nunit];
        float zc = acc[2][mt][e] + xr[2 * UU + nunit];
        float zo = acc[3][mt][e] + xr[3 * UU + nunit];
        float ig = sigf(zi), fg = sigf(zf), og = sigf(zo);
        float cb = tanhf(zc);
        float cn = fg * cst[mt][e] + ig * cb;
        float hn = og * tanhf(cn);
        bool mk = (t < ml[mt][e]);
        float h2 = mk ? hn : hst[mt][e];
        float c2 = mk ? cn : cst[mt][e];
        cst[mt][e] = c2;
        hst[mt][e] = h2;
        hwrite[(size_t)m * UU + nunit] = (bf16)h2;
        out[((size_t)m * TT + t) * H2 + coloff + nunit] = (bf16)h2;
      }
    }

    // inter-workgroup barrier for this direction (double-buffered h => one
    // barrier per step is sufficient)
    __threadfence();
    __syncthreads();
    if (threadIdx.x == 0) {
      int g0 = atomicAdd(gen, 0);
      int arrived = atomicAdd(cnt, 1);
      if (arrived == NWG - 1) {
        atomicExch(cnt, 0);
        atomicAdd(gen, 1);
      } else {
        while (atomicAdd(gen, 0) == g0) { __builtin_amdgcn_s_sleep(1); }
      }
    }
    __syncthreads();
    __threadfence();
  }
}

// ---------------------------------------------------------------------------
// Host-side launcher
// ---------------------------------------------------------------------------
extern "C" void kernel_launch(void* const* d_in, const int* in_sizes, int n_in,
                              void* d_out, int out_size, void* d_ws,
                              size_t ws_size, hipStream_t stream) {
  (void)in_sizes; (void)n_in; (void)out_size; (void)ws_size;

  const float* x       = (const float*)d_in[0];
  const int*   mel_len = (const int*)  d_in[1];
  const float* l1f_Wk  = (const float*)d_in[2];
  const float* l1f_Wr  = (const float*)d_in[3];
  const float* l1f_b   = (const float*)d_in[4];
  const float* l1b_Wk  = (const float*)d_in[5];
  const float* l1b_Wr  = (const float*)d_in[6];
  const float* l1b_b   = (const float*)d_in[7];
  const float* l2f_Wk  = (const float*)d_in[8];
  const float* l2f_Wr  = (const float*)d_in[9];
  const float* l2f_b   = (const float*)d_in[10];
  const float* l2b_Wk  = (const float*)d_in[11];
  const float* l2b_Wr  = (const float*)d_in[12];
  const float* l2b_b   = (const float*)d_in[13];
  const float* proj_W  = (const float*)d_in[14];
  const float* proj_b  = (const float*)d_in[15];
  float* outp = (float*)d_out;

  // -- workspace carve-up (256B aligned) ------------------------------------
  uint8_t* ws = (uint8_t*)d_ws;
  size_t off = 0;
  auto carve = [&](size_t bytes) -> uint8_t* {
    uint8_t* p = ws + off;
    off = (off + bytes + 255) & ~(size_t)255;
    return p;
  };
  bf16* xbf    = (bf16*)carve((size_t)MROWS * DIN * 2);
  bf16* wk1f_p = (bf16*)carve((size_t)DIN * GG * 2);
  bf16* wk1b_p = (bf16*)carve((size_t)DIN * GG * 2);
  bf16* wk2f_p = (bf16*)carve((size_t)H2 * GG * 2);
  bf16* wk2b_p = (bf16*)carve((size_t)H2 * GG * 2);
  bf16* wr1f_p = (bf16*)carve((size_t)UU * GG * 2);
  bf16* wr1b_p = (bf16*)carve((size_t)UU * GG * 2);
  bf16* wr2f_p = (bf16*)carve((size_t)UU * GG * 2);
  bf16* wr2b_p = (bf16*)carve((size_t)UU * GG * 2);
  bf16* proj_p = (bf16*)carve((size_t)H2 * NMEL * 2);
  float* xg0   = (float*)carve((size_t)MROWS * GG * 4);  // time-major [T,B,G]
  float* xg1   = (float*)carve((size_t)MROWS * GG * 4);
  bf16* h1bf   = (bf16*)carve((size_t)MROWS * H2 * 2);
  bf16* h2bf   = (bf16*)carve((size_t)MROWS * H2 * 2);
  bf16* hstate = (bf16*)carve((size_t)2 * 2 * HSTATE * 2);  // 131072 B (256-mult)
  int*  bars   = (int*)carve(64);                            // directly after

  // -- 1) convert inputs / pack weights to bf16 -----------------------------
  {
    long n = (long)MROWS * DIN;
    f32_to_bf16_kernel<<<4096, 256, 0, stream>>>(x, xbf, n);
  }
  auto pack = [&](const float* W, bf16* dst, int K, int N) {
    long n = (long)K * N;
    pack_b_kernel<<<(unsigned)((n + 255) / 256), 256, 0, stream>>>(W, dst, K, N);
  };
  pack(l1f_Wk, wk1f_p, DIN, GG);
  pack(l1b_Wk, wk1b_p, DIN, GG);
  pack(l2f_Wk, wk2f_p, H2, GG);
  pack(l2b_Wk, wk2b_p, H2, GG);
  pack(l1f_Wr, wr1f_p, UU, GG);
  pack(l1b_Wr, wr1b_p, UU, GG);
  pack(l2f_Wr, wr2f_p, UU, GG);
  pack(l2b_Wr, wr2b_p, UU, GG);
  pack(proj_W, proj_p, H2, NMEL);

  // Big GEMMs: 32x64 macro-tiles per wave (NT=4).
  auto gemm4 = [&](const bf16* A, const bf16* Bp, const float* bias, float* C,
                   int M, int K, int N, int remapT) {
    long macros = (long)(M >> 5) * (N >> 6);
    unsigned blocks = (unsigned)((macros + 7) / 8);
    wmma_gemm_blocked_kernel<4><<<blocks, 256, 0, stream>>>(A, Bp, bias, C, M,
                                                            K, N, remapT);
  };
  // Projection GEMM: N=80 -> 32x80 macro-tiles per wave (NT=5).
  auto gemm5 = [&](const bf16* A, const bf16* Bp, const float* bias, float* C,
                   int M, int K, int N, int remapT) {
    long macros = (long)(M >> 5) * (N / 80);
    unsigned blocks = (unsigned)((macros + 7) / 8);
    wmma_gemm_blocked_kernel<5><<<blocks, 256, 0, stream>>>(A, Bp, bias, C, M,
                                                            K, N, remapT);
  };
  auto zero_state = [&]() {
    // hstate is exactly 131072 B (a 256 multiple), bars sits right after:
    // zero both precisely.
    long words = ((long)2 * 2 * HSTATE * 2) / 4 + 16;
    zero_u32_kernel<<<256, 256, 0, stream>>>((unsigned int*)hstate, words);
  };

  // -- 2) layer 1: xg GEMMs + scan ------------------------------------------
  gemm4(xbf, wk1f_p, l1f_b, xg0, MROWS, DIN, GG, TT);
  gemm4(xbf, wk1b_p, l1b_b, xg1, MROWS, DIN, GG, TT);
  zero_state();
  lstm_scan_kernel<<<2 * NWG, SCAN_THREADS, 0, stream>>>(
      xg0, xg1, wr1f_p, wr1b_p, hstate, h1bf, mel_len, bars);

  // -- 3) layer 2: xg GEMMs (reuse buffers) + scan --------------------------
  gemm4(h1bf, wk2f_p, l2f_b, xg0, MROWS, H2, GG, TT);
  gemm4(h1bf, wk2b_p, l2b_b, xg1, MROWS, H2, GG, TT);
  zero_state();
  lstm_scan_kernel<<<2 * NWG, SCAN_THREADS, 0, stream>>>(
      xg0, xg1, wr2f_p, wr2b_p, hstate, h2bf, mel_len, bars);

  // -- 4) output projection --------------------------------------------------
  gemm5(h2bf, proj_p, proj_b, outp, MROWS, H2, NMEL, 0);
}
